// PointNetFeaturePropagation_21715354649485
// MI455X (gfx1250) — matrix operand order repaired and checked by reading, not compiled, not run on hardware
//
#include <hip/hip_runtime.h>
#include <hip/hip_bf16.h>

typedef __attribute__((ext_vector_type(16))) __bf16 v16bf;
typedef __attribute__((ext_vector_type(8)))  float  v8f;

#define Bsz   8
#define Npts  8192
#define Spts  2048
#define D1c   256
#define D2c   256
#define Cin   512

union FragBF { unsigned int u[8]; v16bf v; };

__device__ __forceinline__ unsigned short f2bf(float f) {
  unsigned int x = __float_as_uint(f);
  unsigned int r = (x + 0x7FFFu + ((x >> 16) & 1u)) >> 16;
  return (unsigned short)r;
}
__device__ __forceinline__ float bf2f(unsigned short h) {
  return __uint_as_float(((unsigned int)h) << 16);
}
__device__ __forceinline__ unsigned int packbf(float a, float b) {
  return (unsigned int)f2bf(a) | ((unsigned int)f2bf(b) << 16);
}
__device__ __forceinline__ unsigned int lds_off(const void* p) {
  return (unsigned int)(unsigned long long)p;
}
// gfx1250 async copy global -> LDS, 16 bytes per lane (ASYNCcnt tracked)
__device__ __forceinline__ void async_b128(unsigned int lds_byte,
                                           unsigned long long gaddr) {
  asm volatile("global_load_async_to_lds_b128 %0, %1, off"
               :: "v"(lds_byte), "v"(gaddr) : "memory");
}
__device__ __forceinline__ void wait_async0() {
  asm volatile("s_wait_asynccnt 0x0" ::: "memory");
}

// Activation global layout (uint units), fragment-native:
//   idx = ((b * (C/32) + kb) * Npts + n) * 16 + k2l
// where channel c -> kb = c/32, k2l = (c%32)/2, pair packed (c even, c odd).
__device__ __forceinline__ size_t act_idx(int b, int C, int pair, int n) {
  const int kb  = pair >> 4;
  const int k2l = pair & 15;
  return ((size_t)(b * (C >> 5) + kb) * Npts + n) * 16 + k2l;
}

// ---------------------------------------------------------------------------
// 1) f32 -> bf16 flat convert (weights)
// ---------------------------------------------------------------------------
__global__ void cvt_f32_bf16(const float* __restrict__ in,
                             unsigned short* __restrict__ out, int n) {
  int i = blockIdx.x * 256 + threadIdx.x;
  if (i < n) out[i] = f2bf(in[i]);
}

// ---------------------------------------------------------------------------
// 2) points1 [B,256,N] f32 -> tiled act (pairs 0..127); grid (N/256,128,B)
// ---------------------------------------------------------------------------
__global__ void copy_p1(const float* __restrict__ p1,
                        unsigned int* __restrict__ actp) {
  const int n  = blockIdx.x * 256 + threadIdx.x;
  const int c2 = blockIdx.y;
  const int b  = blockIdx.z;
  const float v0 = p1[((size_t)b * D1c + 2 * c2) * Npts + n];
  const float v1 = p1[((size_t)b * D1c + 2 * c2 + 1) * Npts + n];
  actp[act_idx(b, Cin, c2, n)] = packbf(v0, v1);
}

// ---------------------------------------------------------------------------
// 3) 3-NN inverse-distance interpolation -> tiled act (pairs 128..255)
//    grid: (N/64, B), block: 256 (8 waves, 8 points each)
// ---------------------------------------------------------------------------
__global__ void knn3_interp(const float* __restrict__ xyz1,
                            const float* __restrict__ xyz2,
                            const float* __restrict__ points2,
                            unsigned int* __restrict__ actp) {
  __shared__ float sx[Spts], sy[Spts], sz[Spts];
  const int b  = blockIdx.y;
  const int n0 = blockIdx.x * 64;
  const float* x2 = xyz2 + (size_t)b * 3 * Spts;
  for (int i = threadIdx.x; i < Spts; i += 256) {
    sx[i] = x2[i];
    sy[i] = x2[Spts + i];
    sz[i] = x2[2 * Spts + i];
  }
  __syncthreads();

  const int lane = threadIdx.x & 31;
  const int wave = threadIdx.x >> 5;
  const float* x1 = xyz1 + (size_t)b * 3 * Npts;
  const float* p2 = points2 + (size_t)b * D2c * Spts;

  for (int pi = 0; pi < 8; ++pi) {
    const int n = n0 + wave * 8 + pi;
    const float px = x1[n], py = x1[Npts + n], pz = x1[2 * Npts + n];

    float d0 = 3.0e38f, d1 = 3.0e38f, d2 = 3.0e38f;
    int   i0 = 0, i1 = 0, i2 = 0;
#pragma unroll 4
    for (int it = 0; it < Spts / 32; ++it) {
      const int s = lane + 32 * it;
      const float dx = px - sx[s], dy = py - sy[s], dz = pz - sz[s];
      const float d  = dx * dx + dy * dy + dz * dz;
      if (d < d2) {
        if (d < d0)      { d2 = d1; i2 = i1; d1 = d0; i1 = i0; d0 = d; i0 = s; }
        else if (d < d1) { d2 = d1; i2 = i1; d1 = d;  i1 = s; }
        else             { d2 = d;  i2 = s; }
      }
    }
#pragma unroll
    for (int off = 16; off >= 1; off >>= 1) {
      float e0 = __shfl_xor(d0, off, 32); int j0 = __shfl_xor(i0, off, 32);
      float e1 = __shfl_xor(d1, off, 32); int j1 = __shfl_xor(i1, off, 32);
      float e2 = __shfl_xor(d2, off, 32); int j2 = __shfl_xor(i2, off, 32);
      float m0, m1, m2; int k0, k1, k2;
      if (d0 <= e0) {
        m0 = d0; k0 = i0;
        if (d1 <= e0) { m1 = d1; k1 = i1;
                        m2 = (d2 <= e0) ? d2 : e0; k2 = (d2 <= e0) ? i2 : j0; }
        else          { m1 = e0; k1 = j0;
                        m2 = (d1 <= e1) ? d1 : e1; k2 = (d1 <= e1) ? i1 : j1; }
      } else {
        m0 = e0; k0 = j0;
        if (e1 <= d0) { m1 = e1; k1 = j1;
                        m2 = (e2 <= d0) ? e2 : d0; k2 = (e2 <= d0) ? j2 : i0; }
        else          { m1 = d0; k1 = i0;
                        m2 = (d1 <= e1) ? d1 : e1; k2 = (d1 <= e1) ? i1 : j1; }
      }
      d0 = m0; d1 = m1; d2 = m2; i0 = k0; i1 = k1; i2 = k2;
    }

    const float r0 = 1.0f / (d0 + 1e-8f);
    const float r1 = 1.0f / (d1 + 1e-8f);
    const float r2 = 1.0f / (d2 + 1e-8f);
    const float rs = 1.0f / (r0 + r1 + r2);
    const float w0 = r0 * rs, w1 = r1 * rs, w2 = r2 * rs;

#pragma unroll
    for (int i = 0; i < 4; ++i) {
      const int cp = lane + 32 * i;                 // channel pair in [0,128)
      const float* ra = p2 + (size_t)(2 * cp) * Spts;
      const float* rb = p2 + (size_t)(2 * cp + 1) * Spts;
      const float va = w0 * ra[i0] + w1 * ra[i1] + w2 * ra[i2];
      const float vb = w0 * rb[i0] + w1 * rb[i1] + w2 * rb[i2];
      actp[act_idx(b, Cin, 128 + cp, n)] = packbf(va, vb);
    }
  }
}

// ---------------------------------------------------------------------------
// 4) bf16 WMMA GEMM:  Y[b][o][n] = sum_c W[o][c] * X[b][c][n]
//    X tiled [b][kb][n][k2l]. Async global->LDS (b128), double buffered,
//    padded LDS rows (64B payload in 80B rows) for bank spread.
//    block tile 128(o) x 128(n), BK=32; 8 waves, each 32(o) x 64(n)
// ---------------------------------------------------------------------------
#define BM 128
#define BNT 128
#define BK 32
#define LDSROW 80          // bytes per padded LDS row
#define BUFBYTES (128 * LDSROW)

__global__ void __launch_bounds__(256)
gemm_bf16(const unsigned int* __restrict__ Xf,    // tiled activations
          const unsigned short* __restrict__ Wt,  // [Out][In] bf16
          unsigned short* __restrict__ Y,         // [B][Out][Npts] bf16
          int In, int Out) {
  __shared__ __align__(16) unsigned short As[2][BM][40];   // [o][k] padded
  __shared__ __align__(16) unsigned int   Bq[2][BNT][20];  // [n][k2l] padded

  const int b     = blockIdx.z;
  const int oBase = blockIdx.y * BM;
  const int nBase = blockIdx.x * BNT;
  const int tid   = threadIdx.x;
  const int lane  = tid & 31;
  const int wave  = tid >> 5;
  const int woff  = (wave >> 1) * 32;
  const int noff  = (wave & 1) * 64;
  const int halfA = lane >> 4;
  const int rowA  = lane & 15;

  // ---- per-thread async-copy addresses (advance by constant per K-step) ----
  const int trow = tid >> 2;      // 0..63
  const int tseg = tid & 3;       // 16B segment
  unsigned long long gA0 =
      (unsigned long long)(Wt + (size_t)(oBase + trow) * In) + tseg * 16;
  unsigned long long gA1 = gA0 + (unsigned long long)In * 128;  // +64 rows
  unsigned long long gB0 =
      (unsigned long long)(Xf + ((size_t)b * (In >> 5) * Npts + nBase) * 16) +
      (unsigned long long)tid * 16;
  unsigned long long gB1 = gB0 + 4096;
  const unsigned long long gBstep = (unsigned long long)Npts * 64;  // next kb

  const unsigned int uA0 = lds_off(&As[0][0][0]) + trow * LDSROW + tseg * 16;
  const unsigned int uA1 = uA0 + 64 * LDSROW;
  const unsigned int uB0 = lds_off(&Bq[0][0][0]) + trow * LDSROW + tseg * 16;
  const unsigned int uB1 = uB0 + 64 * LDSROW;

  auto prefetch = [&](int buf) {
    const unsigned int bo = buf ? BUFBYTES : 0;
    async_b128(uA0 + bo, gA0);
    async_b128(uA1 + bo, gA1);
    async_b128(uB0 + bo, gB0);
    async_b128(uB1 + bo, gB1);
  };

  v8f acc[2][4];
  const v8f vzero = {0.f, 0.f, 0.f, 0.f, 0.f, 0.f, 0.f, 0.f};
#pragma unroll
  for (int i = 0; i < 2; ++i)
#pragma unroll
    for (int j = 0; j < 4; ++j) acc[i][j] = vzero;

  const int nsteps = In / BK;   // 16
  prefetch(0);
  for (int ks = 0; ks < nsteps; ++ks) {
    wait_async0();          // my copies for buffer (ks&1) have landed
    __syncthreads();        // everyone's copies landed; prior reads retired
    if (ks + 1 < nsteps) {
      gA0 += 64; gA1 += 64;           // next 32 k-columns of W
      gB0 += gBstep; gB1 += gBstep;   // next kb slab of X
      prefetch((ks + 1) & 1);
    }
    const int cur = ks & 1;

    FragBF afr[2];
#pragma unroll
    for (int t = 0; t < 2; ++t) {
      const int o = woff + t * 16 + rowA;
#pragma unroll
      for (int v = 0; v < 8; ++v) {
        const int kk = (v < 4) ? (halfA * 8 + 2 * v)
                               : (16 + halfA * 8 + 2 * (v - 4));
        afr[t].u[v] = *(const unsigned int*)&As[cur][o][kk];
      }
    }
    FragBF bfr[4];
#pragma unroll
    for (int t = 0; t < 4; ++t) {
      const int n = noff + t * 16 + rowA;
#pragma unroll
      for (int v = 0; v < 8; ++v) bfr[t].u[v] = Bq[cur][n][(v < 4) ? (halfA * 4 + v) : (8 + halfA * 4 + (v - 4))];
    }

#pragma unroll
    for (int i = 0; i < 2; ++i)
#pragma unroll
      for (int j = 0; j < 4; ++j)
        acc[i][j] = __builtin_amdgcn_wmma_f32_16x16x32_bf16(
            false, afr[i].v, false, bfr[j].v, (short)0, acc[i][j], false, false);
  }

  unsigned short* Yb = Y + (size_t)b * Out * Npts;
#pragma unroll
  for (int i = 0; i < 2; ++i)
#pragma unroll
    for (int j = 0; j < 4; ++j)
#pragma unroll
      for (int v = 0; v < 8; ++v) {
        const int o = oBase + woff + i * 16 + v + 8 * halfA;
        const int n = nBase + noff + j * 16 + rowA;
        Yb[(size_t)o * Npts + n] = f2bf(acc[i][j][v]);
      }
}

// ---------------------------------------------------------------------------
// 5) BN stats stage 1: per-channel partial sum/sumsq, fixed chunks
// ---------------------------------------------------------------------------
__global__ void bn_stats1(const unsigned short* __restrict__ y,
                          float* __restrict__ psum, float* __restrict__ psq,
                          int C) {
  const int c     = blockIdx.x;
  const int chunk = blockIdx.y;
  const int b     = chunk >> 2;
  const int n0    = (chunk & 3) * 2048;
  const unsigned short* row = y + ((size_t)b * C + c) * Npts + n0;
  float s = 0.f, q = 0.f;
  for (int i = threadIdx.x; i < 2048; i += 256) {
    const float v = bf2f(row[i]);
    s += v; q += v * v;
  }
  __shared__ float rs[256], rq[256];
  rs[threadIdx.x] = s; rq[threadIdx.x] = q;
  __syncthreads();
  for (int off = 128; off > 0; off >>= 1) {
    if (threadIdx.x < off) {
      rs[threadIdx.x] += rs[threadIdx.x + off];
      rq[threadIdx.x] += rq[threadIdx.x + off];
    }
    __syncthreads();
  }
  if (threadIdx.x == 0) {
    psum[c * 32 + chunk] = rs[0];
    psq[c * 32 + chunk]  = rq[0];
  }
}

// 6) BN stats stage 2
__global__ void bn_stats2(const float* __restrict__ psum,
                          const float* __restrict__ psq,
                          const float* __restrict__ g,
                          const float* __restrict__ beta,
                          float* __restrict__ scale, float* __restrict__ shift,
                          int C) {
  const int c = blockIdx.x * 256 + threadIdx.x;
  if (c >= C) return;
  float s = 0.f, q = 0.f;
#pragma unroll
  for (int i = 0; i < 32; ++i) { s += psum[c * 32 + i]; q += psq[c * 32 + i]; }
  const float invM = 1.0f / (float)(Bsz * Npts);
  const float mean = s * invM;
  const float var  = q * invM - mean * mean;
  const float sc   = g[c] * rsqrtf(var + 1e-5f);
  scale[c] = sc;
  shift[c] = beta[c] - mean * sc;
}

// 7) BN + ReLU -> tiled bf16 act; grid (N/256, C/2, B)
__global__ void bn_relu_bf16(const unsigned short* __restrict__ y,
                             const float* __restrict__ scale,
                             const float* __restrict__ shift,
                             unsigned int* __restrict__ actp, int C) {
  const int n  = blockIdx.x * 256 + threadIdx.x;
  const int c2 = blockIdx.y;
  const int b  = blockIdx.z;
  const int c  = 2 * c2;
  float v0 = bf2f(y[((size_t)b * C + c) * Npts + n]) * scale[c] + shift[c];
  float v1 = bf2f(y[((size_t)b * C + c + 1) * Npts + n]) * scale[c + 1] +
             shift[c + 1];
  actp[act_idx(b, C, c2, n)] = packbf(fmaxf(v0, 0.f), fmaxf(v1, 0.f));
}

// 8) final BN + ReLU -> f32 d_out; grid (N/256, C, B)
__global__ void bn_relu_f32(const unsigned short* __restrict__ y,
                            const float* __restrict__ scale,
                            const float* __restrict__ shift,
                            float* __restrict__ out, int C) {
  const int n = blockIdx.x * 256 + threadIdx.x;
  const int c = blockIdx.y;
  const int b = blockIdx.z;
  const size_t idx = ((size_t)b * C + c) * Npts + n;
  const float v = bf2f(y[idx]) * scale[c] + shift[c];
  out[idx] = fmaxf(v, 0.f);
}

// ---------------------------------------------------------------------------
extern "C" void kernel_launch(void* const* d_in, const int* in_sizes, int n_in,
                              void* d_out, int out_size, void* d_ws,
                              size_t ws_size, hipStream_t stream) {
  (void)in_sizes; (void)n_in; (void)out_size; (void)ws_size;
  const float* xyz1    = (const float*)d_in[0];
  const float* xyz2    = (const float*)d_in[1];
  const float* points1 = (const float*)d_in[2];
  const float* points2 = (const float*)d_in[3];
  const float* w0 = (const float*)d_in[4];
  const float* g0 = (const float*)d_in[6];
  const float* be0 = (const float*)d_in[7];
  const float* w1 = (const float*)d_in[8];
  const float* g1 = (const float*)d_in[10];
  const float* be1 = (const float*)d_in[11];
  const float* w2 = (const float*)d_in[12];
  const float* g2 = (const float*)d_in[14];
  const float* be2 = (const float*)d_in[15];
  // (conv biases cancel exactly through BN mean subtraction; omitted)

  char* p = (char*)d_ws;
  unsigned int*   actA = (unsigned int*)p;   p += (size_t)Bsz * Cin * Npts * 2;
  unsigned short* actY = (unsigned short*)p; p += (size_t)Bsz * Cin * Npts * 2;
  unsigned short* wb0  = (unsigned short*)p; p += 512 * 512 * 2;
  unsigned short* wb1  = (unsigned short*)p; p += 512 * 512 * 2;
  unsigned short* wb2  = (unsigned short*)p; p += 256 * 512 * 2;
  float* psum  = (float*)p; p += 512 * 32 * 4;
  float* psq   = (float*)p; p += 512 * 32 * 4;
  float* scale = (float*)p; p += 512 * 4;
  float* shift = (float*)p; p += 512 * 4;

  cvt_f32_bf16<<<(512 * 512) / 256, 256, 0, stream>>>(w0, wb0, 512 * 512);
  cvt_f32_bf16<<<(512 * 512) / 256, 256, 0, stream>>>(w1, wb1, 512 * 512);
  cvt_f32_bf16<<<(256 * 512) / 256, 256, 0, stream>>>(w2, wb2, 256 * 512);

  copy_p1<<<dim3(Npts / 256, D1c / 2, Bsz), 256, 0, stream>>>(points1, actA);
  knn3_interp<<<dim3(Npts / 64, Bsz), 256, 0, stream>>>(xyz1, xyz2, points2,
                                                        actA);

  // layer 0: 512 -> 512
  gemm_bf16<<<dim3(Npts / BNT, 512 / BM, Bsz), 256, 0, stream>>>(actA, wb0,
                                                                 actY, 512, 512);
  bn_stats1<<<dim3(512, 32), 256, 0, stream>>>(actY, psum, psq, 512);
  bn_stats2<<<2, 256, 0, stream>>>(psum, psq, g0, be0, scale, shift, 512);
  bn_relu_bf16<<<dim3(Npts / 256, 256, Bsz), 256, 0, stream>>>(actY, scale,
                                                               shift, actA, 512);
  // layer 1: 512 -> 512
  gemm_bf16<<<dim3(Npts / BNT, 512 / BM, Bsz), 256, 0, stream>>>(actA, wb1,
                                                                 actY, 512, 512);
  bn_stats1<<<dim3(512, 32), 256, 0, stream>>>(actY, psum, psq, 512);
  bn_stats2<<<2, 256, 0, stream>>>(psum, psq, g1, be1, scale, shift, 512);
  bn_relu_bf16<<<dim3(Npts / 256, 256, Bsz), 256, 0, stream>>>(actY, scale,
                                                               shift, actA, 512);
  // layer 2: 512 -> 256, f32 output
  gemm_bf16<<<dim3(Npts / BNT, 256 / BM, Bsz), 256, 0, stream>>>(actA, wb2,
                                                                 actY, 512, 256);
  bn_stats1<<<dim3(256, 32), 256, 0, stream>>>(actY, psum, psq, 256);
  bn_stats2<<<1, 256, 0, stream>>>(psum, psq, g2, be2, scale, shift, 256);
  bn_relu_f32<<<dim3(Npts / 256, 256, Bsz), 256, 0, stream>>>(
      actY, scale, shift, (float*)d_out, 256);
}
